// DTIHarmonic_76682346102832
// MI455X (gfx1250) — matrix operand order repaired and checked by reading, not compiled, not run on hardware
//
#include <hip/hip_runtime.h>

#define N1_DIM 128
#define N2_DIM 512
#define D_DIM  128
#define L_DIM  3
#define T_DIM  8
#define FIN_DIM 56
#define FPAD_DIM 64
#define H_DIM  128

typedef __attribute__((ext_vector_type(16))) __bf16 v16bf;
typedef __attribute__((ext_vector_type(8)))  float  v8f;

__device__ __forceinline__ uint32_t pack_bf16(float a, float b) {
    union { __bf16 h[2]; uint32_t u; } p;
    p.h[0] = (__bf16)a; p.h[1] = (__bf16)b;
    return p.u;
}
__device__ __forceinline__ uint16_t bf16bits(float a) {
    union { __bf16 h; uint16_t u; } p;
    p.h = (__bf16)a;
    return p.u;
}

// ---------------------------------------------------------------------------
// Zero-pad K dimension: dst[r, 0:srcK] = src[r, :], dst[r, srcK:dstK] = 0
// ---------------------------------------------------------------------------
__global__ void pad_cols_kernel(const float* __restrict__ src, float* __restrict__ dst,
                                int rows, int srcK, int dstK)
{
    int idx = blockIdx.x * blockDim.x + threadIdx.x;
    if (idx >= rows * dstK) return;
    int r = idx / dstK, k = idx % dstK;
    dst[idx] = (k < srcK) ? src[(long long)r * srcK + k] : 0.f;
}

// Zero-pad rows (same column count): dst[0:srcRows,:] = src, rest 0
__global__ void pad_rows_kernel(const float* __restrict__ src, float* __restrict__ dst,
                                int dstRows, int srcRows, int cols)
{
    int idx = blockIdx.x * blockDim.x + threadIdx.x;
    if (idx >= dstRows * cols) return;
    int r = idx / cols;
    dst[idx] = (r < srcRows) ? src[idx] : 0.f;
}

// ---------------------------------------------------------------------------
// Batched GEMM: C[b] = act( A[b] @ B(,b) (+ bias) ), bf16 WMMA, f32 accumulate.
// Block = 128 threads (4 waves) -> one 16(M) x 64(N) strip of C; wave w owns
// the 16x16 tile at N offset w*16. Requires M%16==0, N%64==0, K%32==0 (inputs
// with K=56 are zero-padded to 64 beforehand, so the hot loop is branch-free).
// A/B staged in LDS as packed bf16 so each lane's WMMA fragment is gathered
// with contiguous ds_load_b128s (layouts per CDNA5 ISA §7.12.2).
// ---------------------------------------------------------------------------
template <int TRANSB>
__global__ void gemm_wmma_kernel(const float* __restrict__ A, long long sAb, int lda,
                                 const float* __restrict__ Bm, long long sBb, int ldb,
                                 const float* __restrict__ bias,
                                 float* __restrict__ C, long long sCb, int ldc,
                                 int M, int N, int K, int act)
{
    __shared__ __align__(16) uint32_t sA[16 * 16];   // [row][kp], kp = k/2 (pairs)
    __shared__ __align__(16) uint16_t sB[64 * 32];   // [n][k] bf16, row = 32 bf16 = 64B

    const int tid  = threadIdx.x;
    const int wave = tid >> 5;
    const int lane = tid & 31;
    const int g    = lane >> 4;      // half-wave
    const int lr   = lane & 15;      // A row / B,C col within tile
    const int tileM  = blockIdx.x * 16;
    const int tileN0 = blockIdx.y * 64;
    const int b      = blockIdx.z;

    const float* Ab = A  + (long long)b * sAb + (long long)tileM * lda;
    const float* Bb = Bm + (long long)b * sBb;
    float*       Cb = C  + (long long)b * sCb;

    // Per-thread staging coordinates (invariant over K loop)
    const int arow = tid >> 3;            // 0..15
    const int akc  = (tid & 7) << 2;      // 0,4,...,28

    v8f acc = {0.f, 0.f, 0.f, 0.f, 0.f, 0.f, 0.f, 0.f};

    for (int kk = 0; kk < K; kk += 32) {
        // ---- stage A: 16 rows x 32 k (512 floats, float4 per thread) ------
        {
            float4 f = *(const float4*)(Ab + (long long)arow * lda + kk + akc);
            sA[arow * 16 + (akc >> 1) + 0] = pack_bf16(f.x, f.y);
            sA[arow * 16 + (akc >> 1) + 1] = pack_bf16(f.z, f.w);
        }
        // ---- stage B: 32 k x 64 n -> sB[n][k] (2048 floats, 16/thread) ----
        if (TRANSB) {
            // B stored [n][k] row-major: float4 along k, pack pairs along k.
#pragma unroll
            for (int q = 0; q < 4; ++q) {
                int idx = tid + q * 128;   // 0..511
                int n   = idx >> 3;        // 0..63
                int kc  = (idx & 7) << 2;  // 0..28
                float4 f = *(const float4*)(Bb + (long long)(tileN0 + n) * ldb + kk + kc);
                uint32_t* dst = (uint32_t*)&sB[n * 32 + kc];
                dst[0] = pack_bf16(f.x, f.y);
                dst[1] = pack_bf16(f.z, f.w);
            }
        } else {
            // B stored [k][n] row-major: float4 along n (coalesced), scatter
            // 4 bf16 to 4 LDS rows.
#pragma unroll
            for (int q = 0; q < 4; ++q) {
                int idx = tid + q * 128;   // 0..511
                int k   = idx >> 4;        // 0..31
                int nc  = (idx & 15) << 2; // 0..60
                float4 f = *(const float4*)(Bb + (long long)(kk + k) * ldb + tileN0 + nc);
                sB[(nc + 0) * 32 + k] = bf16bits(f.x);
                sB[(nc + 1) * 32 + k] = bf16bits(f.y);
                sB[(nc + 2) * 32 + k] = bf16bits(f.z);
                sB[(nc + 3) * 32 + k] = bf16bits(f.w);
            }
        }
        __syncthreads();

        // ---- per-lane fragment gather (contiguous b128 LDS reads) + WMMA ---
        // A frag: uints[0..3] = sA[lr][g*4 + 0..3]   (k = g*8   .. g*8+7)
        //         uints[4..7] = sA[lr][8 + g*4 ...]  (k = 16+g*8.. )
        // B frag: same k order as A: uints[0..3] = k g*8.., uints[4..7] = k 16+g*8..
        const uint32_t* sB32 = (const uint32_t*)sB;
        union { uint32_t u[8]; uint4 q[2]; v16bf v; } av, bv;
        const int n = wave * 16 + lr;
        av.q[0] = *(const uint4*)&sA[lr * 16 + g * 4];
        av.q[1] = *(const uint4*)&sA[lr * 16 + 8 + g * 4];
        bv.q[0] = *(const uint4*)&sB32[n * 16 + g * 4];
        bv.q[1] = *(const uint4*)&sB32[n * 16 + 8 + g * 4];

        acc = __builtin_amdgcn_wmma_f32_16x16x32_bf16(false, av.v, false, bv.v,
                                                      (short)0, acc, false, false);
        __syncthreads();
    }

    const int bcol = tileN0 + wave * 16 + lr;
    const float bb = bias ? bias[bcol] : 0.f;
#pragma unroll
    for (int r = 0; r < 8; ++r) {
        int row = tileM + r + (g << 3);     // C/D: VGPR r -> M = r + 8*g
        float v = acc[r] + bb;
        if (act == 1) v = fmaxf(v, 0.f);
        Cb[(long long)row * ldc + bcol] = v;
    }
}

// ---------------------------------------------------------------------------
// att = softmax(where(adj>0, e + e^T, -9e15)) * adj, one row per block
// ---------------------------------------------------------------------------
__global__ void softmax_att_kernel(const float* __restrict__ e,
                                   const float* __restrict__ adj,
                                   float* __restrict__ att, int N)
{
    const int b = blockIdx.y, i = blockIdx.x, tid = threadIdx.x;
    const float* eb   = e   + (long long)b * N * N;
    const float* adjr = adj + ((long long)b * N + i) * N;
    float*       atr  = att + ((long long)b * N + i) * N;

    __shared__ float red[256];
    __shared__ float rowbuf[N2_DIM];

    float lmax = -3.0e38f;
    for (int j = tid; j < N; j += 256) {
        float v = eb[(long long)i * N + j] + eb[(long long)j * N + i];
        v = (adjr[j] > 0.f) ? v : -9.0e15f;
        rowbuf[j] = v;
        lmax = fmaxf(lmax, v);
    }
    red[tid] = lmax; __syncthreads();
    for (int s = 128; s > 0; s >>= 1) { if (tid < s) red[tid] = fmaxf(red[tid], red[tid + s]); __syncthreads(); }
    float mx = red[0]; __syncthreads();

    float lsum = 0.f;
    for (int j = tid; j < N; j += 256) {
        float ex = __expf(rowbuf[j] - mx);
        rowbuf[j] = ex;
        lsum += ex;
    }
    red[tid] = lsum; __syncthreads();
    for (int s = 128; s > 0; s >>= 1) { if (tid < s) red[tid] += red[tid + s]; __syncthreads(); }
    float inv = 1.0f / red[0];
    for (int j = tid; j < N; j += 256) atr[j] = rowbuf[j] * inv * adjr[j];
}

// ---------------------------------------------------------------------------
// x = c*x + (1-c)*hp, c = sigmoid([x,hp] @ gW + gb), one node per thread
// ---------------------------------------------------------------------------
__global__ void gate_kernel(float* __restrict__ x, const float* __restrict__ hp,
                            const float* __restrict__ gW, const float* __restrict__ gb,
                            int N, int D)
{
    const int b = blockIdx.y;
    const int node = blockIdx.x * blockDim.x + threadIdx.x;
    if (node >= N) return;
    float* xr = x + ((long long)b * N + node) * D;
    const float* hr = hp + ((long long)b * N + node) * D;
    float s = gb[0];
    for (int d = 0; d < D; ++d) s += xr[d] * gW[d];
    for (int d = 0; d < D; ++d) s += hr[d] * gW[D + d];
    float c = 1.f / (1.f + __expf(-s));
    for (int d = 0; d < D; ++d) xr[d] = c * xr[d] + (1.f - c) * hr[d];
}

// ---------------------------------------------------------------------------
// Fused pairwise energies. One block per (b,i); deterministic block partials.
// ---------------------------------------------------------------------------
__global__ void pairwise_kernel(const float* __restrict__ P1, const float* __restrict__ P2,
                                const float* __restrict__ W2A, const float* __restrict__ b2A,
                                const float* __restrict__ W2B, const float* __restrict__ b2B,
                                const float* __restrict__ dmv, const float* __restrict__ A_int,
                                const float* __restrict__ Cp,
                                const float* __restrict__ q1, const float* __restrict__ q2,
                                const float* __restrict__ vr1, const float* __restrict__ vr2,
                                const float* __restrict__ val1, const float* __restrict__ val2,
                                const float* __restrict__ nm1, const float* __restrict__ nm2,
                                const float* __restrict__ ccoef, const float* __restrict__ vcoef,
                                float* __restrict__ part, int Bn, int N1c, int N2c, int Hc)
{
    const int b = blockIdx.y, i = blockIdx.x, tid = threadIdx.x;
    __shared__ float sP1[16 * H_DIM];
    __shared__ float sW2[16 * H_DIM];
    __shared__ float sred[256];

    for (int idx = tid; idx < 16 * H_DIM; idx += 256) {
        int s = idx >> 7, h = idx & 127;
        sP1[idx] = P1[(((long long)s * Bn + b) * N1c + i) * Hc + h];
        int t = s >> 1;
        sW2[idx] = (s & 1) ? W2B[t * Hc + h] : W2A[t * Hc + h];
    }
    __syncthreads();

    const float inv4s2[8] = {
        1.f/(4.f*1.159f*1.159f), 1.f/(4.f*0.448f*0.448f),
        1.f/(4.f*0.927f*0.927f), 1.f/(4.f*0.902f*0.902f),
        1.f/(4.f*0.349f*0.349f), 1.f/(4.f*0.789f*0.789f),
        1.f/(4.f*0.198f*0.198f), 1.f/(4.f*0.317f*0.317f)};
    float Cl[8], b2l[16];
    for (int t = 0; t < 8; ++t) Cl[t] = Cp[t];
    for (int t = 0; t < 8; ++t) { b2l[2*t] = b2A[t]; b2l[2*t+1] = b2B[t]; }

    const float qi  = q1[b * N1c + i];
    const float vri = vr1[b * N1c + i];
    const float v1  = val1[b * N1c + i];
    const float m1  = nm1[b * N1c + i];
    const float cck = ccoef[0], vck = vcoef[0];

    float accT[8] = {0,0,0,0,0,0,0,0};
    float accC = 0.f, accV = 0.f;

    for (int j = tid; j < N2c; j += 256) {
        const float* dv = dmv + (((long long)b * N1c + i) * N2c + j) * 3;
        float dx = dv[0], dy = dv[1], dz = dv[2];
        float dm = sqrtf(dx*dx + dy*dy + dz*dz + 1e-10f);
        if (dm < 0.5f) dm = 1e10f;
        float invdm = 1.f / dm;

        float sv[16];
        for (int s = 0; s < 16; ++s) {
            const float* p2 = P2 + (((long long)s * Bn + b) * N2c + j) * Hc;
            const float* p1 = sP1 + s * H_DIM;
            const float* w2 = sW2 + s * H_DIM;
            float dot = 0.f;
#pragma unroll 4
            for (int h = 0; h < H_DIM; h += 4) {
                float4 a = *(const float4*)(p2 + h);
                dot += fmaxf(p1[h+0] + a.x, 0.f) * w2[h+0];
                dot += fmaxf(p1[h+1] + a.y, 0.f) * w2[h+1];
                dot += fmaxf(p1[h+2] + a.z, 0.f) * w2[h+2];
                dot += fmaxf(p1[h+3] + a.w, 0.f) * w2[h+3];
            }
            sv[s] = 1.f / (1.f + __expf(-(dot + b2l[s])));
        }
#pragma unroll
        for (int t = 0; t < 8; ++t) {
            float Ai = sv[2*t] * 4.f + (t == 1 ? 1.f : 0.f);
            float Bi = (sv[2*t+1] * 3.f + 1.f) * inv4s2[t];
            float d  = dm - Cl[t];
            float ai = A_int[(((long long)b * T_DIM + t) * N1c + i) * N2c + j];
            accT[t] += fminf(Ai * (Bi * d * d - 1.f) * ai, 0.f);
        }
        float ec = cck * qi * q2[b * N2c + j] * invdm * v1 * val2[b * N2c + j];
        accC += fminf(fmaxf(ec, -100.f), 100.f);
        float r = (vri + vr2[b * N2c + j]) * invdm;
        float r2 = r * r, r4 = r2 * r2, r8 = r4 * r4;
        accV += fminf(vck * (r8 - 2.f * r4) * m1 * nm2[b * N2c + j], 100.f);
    }

    for (int k = 0; k < 10; ++k) {
        float v = (k < 8) ? accT[k] : (k == 8 ? accC : accV);
        sred[tid] = v; __syncthreads();
        for (int st = 128; st > 0; st >>= 1) { if (tid < st) sred[tid] += sred[tid + st]; __syncthreads(); }
        if (tid == 0) part[((long long)b * N1c + i) * 10 + k] = sred[0];
        __syncthreads();
    }
}

// ---------------------------------------------------------------------------
// Final [B,13] assembly: 8 harmonics, sasa, rotor, coulomb, vdw, intercept
// ---------------------------------------------------------------------------
__global__ void finalize_kernel(const float* __restrict__ part,
                                const float* __restrict__ sasa, const float* __restrict__ rotor,
                                const float* __restrict__ sasa_c, const float* __restrict__ rotor_c,
                                const float* __restrict__ intercept,
                                float* __restrict__ out, int Bn, int N1c)
{
    int idx = blockIdx.x * blockDim.x + threadIdx.x;
    if (idx >= Bn * 13) return;
    int b = idx / 13, k = idx % 13;
    float v;
    if (k < 8 || k == 10 || k == 11) {
        int kk = (k < 8) ? k : (k - 2);
        float s = 0.f;
        for (int i = 0; i < N1c; ++i) s += part[((long long)b * N1c + i) * 10 + kk];
        v = s;
    } else if (k == 8) v = fminf(sasa_c[0] * sasa[b], 0.f);
    else if (k == 9)   v = fmaxf(rotor_c[0] * rotor[b], 0.f);
    else               v = intercept[0];
    out[idx] = v;
}

// ---------------------------------------------------------------------------
static inline void launch_gemm(hipStream_t st,
                               const float* A, long long sAb, int lda,
                               const float* Bm, long long sBb, int ldb, int transB,
                               const float* bias, float* C, long long sCb, int ldc,
                               int M, int N, int K, int act, int Bn)
{
    dim3 grid(M / 16, N / 64, Bn);
    if (transB)
        gemm_wmma_kernel<1><<<grid, dim3(128, 1, 1), 0, st>>>(A, sAb, lda, Bm, sBb, ldb,
                                                              bias, C, sCb, ldc, M, N, K, act);
    else
        gemm_wmma_kernel<0><<<grid, dim3(128, 1, 1), 0, st>>>(A, sAb, lda, Bm, sBb, ldb,
                                                              bias, C, sCb, ldc, M, N, K, act);
}

extern "C" void kernel_launch(void* const* d_in, const int* in_sizes, int n_in,
                              void* d_out, int out_size, void* d_ws, size_t ws_size,
                              hipStream_t stream)
{
    (void)n_in; (void)out_size; (void)ws_size;
    const float* h1    = (const float*)d_in[0];
    const float* adj1  = (const float*)d_in[1];
    const float* h2    = (const float*)d_in[2];
    const float* adj2  = (const float*)d_in[3];
    const float* A_int = (const float*)d_in[4];
    const float* dmv   = (const float*)d_in[5];
    const float* sasa  = (const float*)d_in[6];
    const float* rotor = (const float*)d_in[8];
    const float* q1    = (const float*)d_in[9];
    const float* q2    = (const float*)d_in[10];
    const float* vr1   = (const float*)d_in[11];
    const float* vr2   = (const float*)d_in[12];
    const float* val1  = (const float*)d_in[13];
    const float* val2  = (const float*)d_in[14];
    const float* nm1   = (const float*)d_in[15];
    const float* nm2   = (const float*)d_in[16];
    const float* emb_W = (const float*)d_in[17];
    const float* gat_W = (const float*)d_in[18];
    const float* gat_Wb= (const float*)d_in[19];
    const float* gat_A = (const float*)d_in[20];
    const float* gat_gW= (const float*)d_in[21];
    const float* gat_gb= (const float*)d_in[22];
    const float* cA_W1 = (const float*)d_in[23];
    const float* cA_b1 = (const float*)d_in[24];
    const float* cA_W2 = (const float*)d_in[25];
    const float* cA_b2 = (const float*)d_in[26];
    const float* cB_W1 = (const float*)d_in[27];
    const float* cB_b1 = (const float*)d_in[28];
    const float* cB_W2 = (const float*)d_in[29];
    const float* cB_b2 = (const float*)d_in[30];
    const float* C_par = (const float*)d_in[31];
    const float* sasa_c= (const float*)d_in[32];
    const float* rotor_c=(const float*)d_in[33];
    const float* vdw_c = (const float*)d_in[34];
    const float* cou_c = (const float*)d_in[35];
    const float* icept = (const float*)d_in[36];

    const int Bn = in_sizes[6];           // sasa has B elements
    const int N1 = N1_DIM, N2 = N2_DIM, D = D_DIM, H = H_DIM;
    const int FIN = FIN_DIM, FP = FPAD_DIM;

    float* w = (float*)d_ws;
    size_t o = 0;
    float* x1   = w + o; o += (size_t)Bn * N1 * D;
    float* x2   = w + o; o += (size_t)Bn * N2 * D;
    float* hb1  = w + o; o += (size_t)Bn * N1 * D;
    float* hb2  = w + o; o += (size_t)Bn * N2 * D;
    float* hA1  = w + o; o += (size_t)Bn * N1 * D;
    float* hA2  = w + o; o += (size_t)Bn * N2 * D;
    float* e1   = w + o; o += (size_t)Bn * N1 * N1;
    float* e2   = w + o; o += (size_t)Bn * N2 * N2;
    float* at1  = w + o; o += (size_t)Bn * N1 * N1;
    float* at2  = w + o; o += (size_t)Bn * N2 * N2;
    float* hp1  = w + o; o += (size_t)Bn * N1 * D;
    float* hp2  = w + o; o += (size_t)Bn * N2 * D;
    float* P1   = w + o; o += (size_t)16 * Bn * N1 * H;
    float* P2   = w + o; o += (size_t)16 * Bn * N2 * H;
    float* part = w + o; o += (size_t)Bn * N1 * 10;
    float* h1p  = w + o; o += (size_t)Bn * N1 * FP;   // K-padded copies (K=56->64)
    float* h2p  = w + o; o += (size_t)Bn * N2 * FP;
    float* embp = w + o; o += (size_t)FP * D;

    // Zero-pad the K=56 operands so every GEMM has K % 32 == 0 (branch-free
    // WMMA hot loop).
    {
        int n1 = Bn * N1 * FP;
        pad_cols_kernel<<<(n1 + 255) / 256, 256, 0, stream>>>(h1, h1p, Bn * N1, FIN, FP);
        int n2 = Bn * N2 * FP;
        pad_cols_kernel<<<(n2 + 255) / 256, 256, 0, stream>>>(h2, h2p, Bn * N2, FIN, FP);
        int n3 = FP * D;
        pad_rows_kernel<<<(n3 + 255) / 256, 256, 0, stream>>>(emb_W, embp, FP, FIN, D);
    }

    // Embedding: x = h_pad @ emb_W_pad (K=64)
    launch_gemm(stream, h1p, (long long)N1*FP, FP, embp, 0, D, 0, nullptr,
                x1, (long long)N1*D, D, N1, D, FP, 0, Bn);
    launch_gemm(stream, h2p, (long long)N2*FP, FP, embp, 0, D, 0, nullptr,
                x2, (long long)N2*D, D, N2, D, FP, 0, Bn);

    // Shared GAT layers on both graphs
    for (int l = 0; l < L_DIM; ++l) {
        const float* Wl  = gat_W  + (size_t)l * D * D;
        const float* Wbl = gat_Wb + (size_t)l * D;
        const float* Al  = gat_A  + (size_t)l * D * D;
        const float* gWl = gat_gW + (size_t)l * 2 * D;
        const float* gbl = gat_gb + l;
        // graph 1 (N1)
        launch_gemm(stream, x1, (long long)N1*D, D, Wl, 0, D, 0, Wbl, hb1, (long long)N1*D, D, N1, D, D, 0, Bn);
        launch_gemm(stream, hb1,(long long)N1*D, D, Al, 0, D, 0, nullptr, hA1,(long long)N1*D, D, N1, D, D, 0, Bn);
        launch_gemm(stream, hA1,(long long)N1*D, D, hb1,(long long)N1*D, D, 1, nullptr, e1,(long long)N1*N1, N1, N1, N1, D, 0, Bn);
        softmax_att_kernel<<<dim3(N1, Bn), 256, 0, stream>>>(e1, adj1, at1, N1);
        launch_gemm(stream, at1,(long long)N1*N1, N1, hb1,(long long)N1*D, D, 0, nullptr, hp1,(long long)N1*D, D, N1, D, N1, 1, Bn);
        gate_kernel<<<dim3((N1+127)/128, Bn), 128, 0, stream>>>(x1, hp1, gWl, gbl, N1, D);
        // graph 2 (N2)
        launch_gemm(stream, x2, (long long)N2*D, D, Wl, 0, D, 0, Wbl, hb2, (long long)N2*D, D, N2, D, D, 0, Bn);
        launch_gemm(stream, hb2,(long long)N2*D, D, Al, 0, D, 0, nullptr, hA2,(long long)N2*D, D, N2, D, D, 0, Bn);
        launch_gemm(stream, hA2,(long long)N2*D, D, hb2,(long long)N2*D, D, 1, nullptr, e2,(long long)N2*N2, N2, N2, N2, D, 0, Bn);
        softmax_att_kernel<<<dim3(N2, Bn), 256, 0, stream>>>(e2, adj2, at2, N2);
        launch_gemm(stream, at2,(long long)N2*N2, N2, hb2,(long long)N2*D, D, 0, nullptr, hp2,(long long)N2*D, D, N2, D, N2, 1, Bn);
        gate_kernel<<<dim3((N2+127)/128, Bn), 128, 0, stream>>>(x2, hp2, gWl, gbl, N2, D);
    }

    // Factorized pair-MLP layer-1: P1 = x1 @ W1_top, P2 = x2 @ W1_bot + b1
    for (int t = 0; t < T_DIM; ++t) {
        for (int m = 0; m < 2; ++m) {
            const float* W1 = (m ? cB_W1 : cA_W1) + (size_t)t * 2 * D * H;
            const float* b1 = (m ? cB_b1 : cA_b1) + (size_t)t * H;
            int s = t * 2 + m;
            launch_gemm(stream, x1, (long long)N1*D, D, W1, 0, H, 0, nullptr,
                        P1 + (size_t)s * Bn * N1 * H, (long long)N1*H, H, N1, H, D, 0, Bn);
            launch_gemm(stream, x2, (long long)N2*D, D, W1 + (size_t)D * H, 0, H, 0, b1,
                        P2 + (size_t)s * Bn * N2 * H, (long long)N2*H, H, N2, H, D, 0, Bn);
        }
    }

    pairwise_kernel<<<dim3(N1, Bn), 256, 0, stream>>>(
        P1, P2, cA_W2, cA_b2, cB_W2, cB_b2, dmv, A_int, C_par,
        q1, q2, vr1, vr2, val1, val2, nm1, nm2, cou_c, vdw_c,
        part, Bn, N1, N2, H);

    finalize_kernel<<<1, 64, 0, stream>>>(part, sasa, rotor, sasa_c, rotor_c, icept,
                                          (float*)d_out, Bn, N1);
}
